// Res_NL_42915313221789
// MI455X (gfx1250) — matrix operand order repaired and checked by reading, not compiled, not run on hardware
//
#include <hip/hip_runtime.h>
#include <hip/hip_bf16.h>
#include <stdint.h>

#define HH 128
#define WW 128
#define HWPIX (HH * WW)
#define CPAD 64          // padded channel count for WMMA K dimension
#define CVALID 42        // real channel count in residual trunk

typedef __attribute__((ext_vector_type(16))) __bf16 v16bf;
typedef __attribute__((ext_vector_type(8)))  float  v8f;

union Frag16 { v16bf v; uint4 q[2]; };
union Frag8  { v8f   v; float f[8]; };

// ---------------------------------------------------------------------------
// Small direct fp32 kernels (latency-bound parts of the net)
// ---------------------------------------------------------------------------

// 3x3 conv, NCHW in -> NCHW out, no bias
__global__ void conv3_nchw_k(const float* __restrict__ in, const float* __restrict__ w,
                             float* __restrict__ out, int ci, int co)
{
    int tid = blockIdx.x * blockDim.x + threadIdx.x;
    if (tid >= co * HWPIX) return;
    int oc = tid / HWPIX;
    int pix = tid - oc * HWPIX;
    int y = pix >> 7, x = pix & (WW - 1);
    float s = 0.f;
    for (int ic = 0; ic < ci; ++ic)
        for (int ky = 0; ky < 3; ++ky) {
            int yy = y + ky - 1;
            if (yy < 0 || yy >= HH) continue;
            for (int kx = 0; kx < 3; ++kx) {
                int xx = x + kx - 1;
                if (xx < 0 || xx >= WW) continue;
                s += in[ic * HWPIX + yy * WW + xx] * w[((oc * ci + ic) * 3 + ky) * 3 + kx];
            }
        }
    out[oc * HWPIX + pix] = s;
}

// 3x3 conv, NCHW in -> NHWC64 trunk tensor at channel offset chbase
__global__ void conv3_to_x_k(const float* __restrict__ in, const float* __restrict__ w,
                             float* __restrict__ xf, int ci, int co, int chbase)
{
    int tid = blockIdx.x * blockDim.x + threadIdx.x;
    if (tid >= co * HWPIX) return;
    int oc = tid / HWPIX;
    int pix = tid - oc * HWPIX;
    int y = pix >> 7, x = pix & (WW - 1);
    float s = 0.f;
    for (int ic = 0; ic < ci; ++ic)
        for (int ky = 0; ky < 3; ++ky) {
            int yy = y + ky - 1;
            if (yy < 0 || yy >= HH) continue;
            for (int kx = 0; kx < 3; ++kx) {
                int xx = x + kx - 1;
                if (xx < 0 || xx >= WW) continue;
                s += in[ic * HWPIX + yy * WW + xx] * w[((oc * ci + ic) * 3 + ky) * 3 + kx];
            }
        }
    xf[(size_t)pix * CPAD + chbase + oc] = s;
}

// 1x1 conv / channel matmul: out[oc] = W[oc,ic] @ in[ic]
__global__ void mat1x1_k(const float* __restrict__ w, const float* __restrict__ in,
                         float* __restrict__ out, int ci, int co)
{
    int tid = blockIdx.x * blockDim.x + threadIdx.x;
    if (tid >= co * HWPIX) return;
    int oc = tid / HWPIX;
    int pix = tid - oc * HWPIX;
    float s = 0.f;
    for (int ic = 0; ic < ci; ++ic)
        s += w[oc * ci + ic] * in[ic * HWPIX + pix];
    out[oc * HWPIX + pix] = s;
}

// initialize attention channels (0..4) of trunk with mlp bias
__global__ void att_bias_k(float* __restrict__ xf, const float* __restrict__ mlp_b)
{
    int tid = blockIdx.x * blockDim.x + threadIdx.x;
    if (tid >= HWPIX * 5) return;
    int pix = tid / 5, c = tid - pix * 5;
    xf[(size_t)pix * CPAD + c] = mlp_b[0];
}

// Windowed neighborhood attention head. theta/phi: C channels, g: 5 channels.
// Accumulates mlp_w[widx] * head_out into trunk channels 0..4.
template <int C, int P>
__global__ void head_k(const float* __restrict__ th, const float* __restrict__ ph,
                       const float* __restrict__ gg, const float* __restrict__ mlp_w,
                       int widx, float* __restrict__ xf)
{
    int pix = blockIdx.x * blockDim.x + threadIdx.x;
    if (pix >= HWPIX) return;
    int y = pix >> 7, x = pix & (WW - 1);
    const float wgt = mlp_w[widx];

    // register-resident theta patch (D = C*P*P <= 72), zero padded
    float tp[C * P * P];
    {
        int d = 0;
#pragma unroll
        for (int ci = 0; ci < C; ++ci)
#pragma unroll
            for (int py = 0; py < P; ++py)
#pragma unroll
                for (int px = 0; px < P; ++px) {
                    int yy = y + py - P / 2, xp = x + px - P / 2;
                    tp[d++] = (yy >= 0 && yy < HH && xp >= 0 && xp < WW)
                                  ? th[ci * HWPIX + yy * WW + xp] : 0.f;
                }
    }

    float lg[49];
    float mx = -1e30f;
    for (int k = 0; k < 49; ++k) {
        int oy = k / 7 - 3, ox = k % 7 - 3;
        int cy = y + oy, cx = x + ox;
        float s = 0.f;
        if (cy >= 0 && cy < HH && cx >= 0 && cx < WW) {
            int d = 0;
#pragma unroll
            for (int ci = 0; ci < C; ++ci)
#pragma unroll
                for (int py = 0; py < P; ++py)
#pragma unroll
                    for (int px = 0; px < P; ++px) {
                        int yy = cy + py - P / 2, xp = cx + px - P / 2;
                        float pv = (yy >= 0 && yy < HH && xp >= 0 && xp < WW)
                                       ? ph[ci * HWPIX + yy * WW + xp] : 0.f;
                        s += tp[d++] * pv;
                    }
        }
        lg[k] = s;
        mx = fmaxf(mx, s);
    }
    float sum = 0.f;
    for (int k = 0; k < 49; ++k) { float e = __expf(lg[k] - mx); lg[k] = e; sum += e; }
    float inv = 1.f / sum;

    float o0 = 0.f, o1 = 0.f, o2 = 0.f, o3 = 0.f, o4 = 0.f;
    for (int k = 0; k < 49; ++k) {
        int oy = k / 7 - 3, ox = k % 7 - 3;
        int cy = y + oy, cx = x + ox;
        if (cy < 0 || cy >= HH || cx < 0 || cx >= WW) continue;
        float a = lg[k] * inv;
        int b = cy * WW + cx;
        o0 += a * gg[0 * HWPIX + b];
        o1 += a * gg[1 * HWPIX + b];
        o2 += a * gg[2 * HWPIX + b];
        o3 += a * gg[3 * HWPIX + b];
        o4 += a * gg[4 * HWPIX + b];
    }
    float* px4 = xf + (size_t)pix * CPAD;
    px4[0] += wgt * o0; px4[1] += wgt * o1; px4[2] += wgt * o2;
    px4[3] += wgt * o3; px4[4] += wgt * o4;
}

// ---------------------------------------------------------------------------
// WMMA residual-block convs (implicit GEMM, bf16 in / f32 accumulate)
// ---------------------------------------------------------------------------

// pack fp32 NHWC64 -> bf16 NHWC64
__global__ void pack_x_k(const float* __restrict__ xf, __bf16* __restrict__ xb)
{
    int i = blockIdx.x * blockDim.x + threadIdx.x;
    if (i >= HWPIX * CPAD) return;
    xb[i] = (__bf16)xf[i];
}

// Pack OIHW (42,42,3,3) fp32 weights into the per-lane B-fragment layout:
//   element index = (((tap*3 + nt)*2 + kc)*32 + lane)*16 + slot
//   n (out ch) = nt*16 + (lane & 15)
//   k (in ch)  = kc*32 + (lane>>4)*16 + slot   (contiguous K order, as in A: lanes 0-15
//   carry K 0..15 of the chunk, lanes 16-31 carry K 16..31)
__global__ void pack_w_k(const float* __restrict__ w, __bf16* __restrict__ wpk)
{
    int i = blockIdx.x * blockDim.x + threadIdx.x;
    if (i >= 9 * 3 * 2 * 32 * 16) return;
    int slot = i & 15;
    int lane = (i >> 4) & 31;
    int kc   = (i >> 9) & 1;
    int nt   = (i >> 10) % 3;
    int tap  = i / 3072;
    int n = nt * 16 + (lane & 15);
    int k = kc * 32 + (lane >> 4) * 16 + slot;
    float v = 0.f;
    if (n < CVALID && k < CVALID)
        v = w[(size_t)(n * CVALID + k) * 9 + tap];
    wpk[i] = (__bf16)v;
}

__device__ __forceinline__ void emit_out(float v, int oc, size_t rowidx,
                                         const float* __restrict__ bias,
                                         const float* __restrict__ resid,
                                         __bf16* __restrict__ obf,
                                         float* __restrict__ of32)
{
    v += bias[oc];
    size_t idx = rowidx + oc;
    if (resid) {
        v += resid[idx];
        v = fmaxf(v, 0.f);
        of32[idx] = v;
        obf[idx] = (__bf16)v;
    } else {
        v = fmaxf(v, 0.f);
        obf[idx] = (__bf16)v;
    }
}

// 4 waves (128 threads) per 64-pixel strip of one image row; 256 blocks.
// Stage: 3 rows x 66 cols x 64 ch of bf16 activations into LDS (zero-padded
// halo), then each wave computes a 16-pixel x 42-outchannel tile with
// v_wmma_f32_16x16x32_bf16, A fragments fed from LDS (branch-free, EXEC
// stays all-ones through the matrix loop as the ISA requires).
#define STRIP 64
#define SCOLS 66                       // STRIP + 2 halo columns

__global__ void __launch_bounds__(128)
wmma_conv_k(const __bf16* __restrict__ in,   // NHWC64 bf16
            const __bf16* __restrict__ wpk,  // packed [9][3][2][32][16]
            const float* __restrict__ bias,  // [42]
            const float* __restrict__ resid, // NHWC64 f32 or null
            __bf16* __restrict__ out_bf,     // NHWC64 bf16
            float* __restrict__ out_f32)     // NHWC64 f32 or null
{
    __shared__ alignas(16) __bf16 smem[3 * SCOLS * CPAD];   // 25,344 B

    int y  = blockIdx.x >> 1;                // image row
    int xs = (blockIdx.x & 1) * STRIP;       // strip base column
    int tid = threadIdx.x;
    int wv = tid >> 5;                       // wave id 0..3 -> pixel tile
    int lane = tid & 31;
    int m = lane & 15;                       // pixel within tile (A-matrix M)
    int kh = lane >> 4;                      // K half select (contiguous K order)

    // hint the packed weights into cache (global_prefetch_b8)
    if (tid < 108)
        __builtin_prefetch(wpk + (size_t)tid * 256, 0, 0);

    // ---- cooperative staging: 3 rows x 66 cols, 128 B per (row,col) ----
    for (int p = tid; p < 3 * SCOLS; p += 128) {
        int ry  = p / SCOLS;                 // 0..2
        int col = p - ry * SCOLS;            // 0..65
        int yy = y + ry - 1;
        int xx = xs + col - 1;
        uint4* dst = (uint4*)(smem + (size_t)p * CPAD);
        if (yy >= 0 && yy < HH && xx >= 0 && xx < WW) {
            const uint4* src = (const uint4*)(in + (size_t)(yy * WW + xx) * CPAD);
#pragma unroll
            for (int q = 0; q < 8; ++q) dst[q] = src[q];
        } else {
#pragma unroll
            for (int q = 0; q < 8; ++q) dst[q] = make_uint4(0u, 0u, 0u, 0u);
        }
    }
    __syncthreads();

    v8f acc0 = {0.f, 0.f, 0.f, 0.f, 0.f, 0.f, 0.f, 0.f};
    v8f acc1 = {0.f, 0.f, 0.f, 0.f, 0.f, 0.f, 0.f, 0.f};
    v8f acc2 = {0.f, 0.f, 0.f, 0.f, 0.f, 0.f, 0.f, 0.f};

#pragma unroll
    for (int ky = 0; ky < 3; ++ky) {
#pragma unroll
        for (int kx = 0; kx < 3; ++kx) {
            int tap = ky * 3 + kx;
            // smem column for this lane's pixel: wv*16 + m + kx  (halo shift folded in)
            const __bf16* arow = smem + (size_t)((ky * SCOLS) + (wv * 16 + m + kx)) * CPAD;
            const __bf16* wb = wpk + (size_t)tap * 3072;
#pragma unroll
            for (int kc = 0; kc < 2; ++kc) {
                Frag16 a;
                {   // lane<16: K {0..15} of chunk; lane>=16: K {16..31}
                    const uint4* p = (const uint4*)(arow + kc * 32 + kh * 16);
                    a.q[0] = p[0];
                    a.q[1] = p[1];
                }
                Frag16 b0, b1, b2;
                const uint4* w0 = (const uint4*)(wb + 0    + kc * 512 + lane * 16);
                const uint4* w1 = (const uint4*)(wb + 1024 + kc * 512 + lane * 16);
                const uint4* w2 = (const uint4*)(wb + 2048 + kc * 512 + lane * 16);
                b0.q[0] = w0[0]; b0.q[1] = w0[1];
                b1.q[0] = w1[0]; b1.q[1] = w1[1];
                b2.q[0] = w2[0]; b2.q[1] = w2[1];
                acc0 = __builtin_amdgcn_wmma_f32_16x16x32_bf16(
                    false, a.v, false, b0.v, (short)0, acc0, false, false);
                acc1 = __builtin_amdgcn_wmma_f32_16x16x32_bf16(
                    false, a.v, false, b1.v, (short)0, acc1, false, false);
                acc2 = __builtin_amdgcn_wmma_f32_16x16x32_bf16(
                    false, a.v, false, b2.v, (short)0, acc2, false, false);
            }
        }
    }

    // C/D layout: lane l holds N = l&15, VGPR r holds M = r + 8*(l>>4)
    Frag8 c0, c1, c2;
    c0.v = acc0; c1.v = acc1; c2.v = acc2;
    int mbase = kh * 8;
    int nl = lane & 15;
#pragma unroll
    for (int r = 0; r < 8; ++r) {
        int pixm = mbase + r;
        size_t rowidx = ((size_t)y * WW + xs + wv * 16 + pixm) * CPAD;
        emit_out(c0.f[r], nl,      rowidx, bias, resid, out_bf, out_f32);
        emit_out(c1.f[r], 16 + nl, rowidx, bias, resid, out_bf, out_f32);
        if (32 + nl < CVALID)
            emit_out(c2.f[r], 32 + nl, rowidx, bias, resid, out_bf, out_f32);
    }
}

// final 42->8 3x3 conv reading NHWC64 trunk, writing NCHW fp32 output
__global__ void recon_k(const float* __restrict__ xf, const float* __restrict__ w,
                        float* __restrict__ out)
{
    int tid = blockIdx.x * blockDim.x + threadIdx.x;
    if (tid >= 8 * HWPIX) return;
    int oc = tid / HWPIX;
    int pix = tid - oc * HWPIX;
    int y = pix >> 7, x = pix & (WW - 1);
    float s = 0.f;
    for (int ky = 0; ky < 3; ++ky) {
        int yy = y + ky - 1;
        if (yy < 0 || yy >= HH) continue;
        for (int kx = 0; kx < 3; ++kx) {
            int xx = x + kx - 1;
            if (xx < 0 || xx >= WW) continue;
            const float* row = xf + (size_t)(yy * WW + xx) * CPAD;
            const float* wk = w + (size_t)oc * CVALID * 9 + ky * 3 + kx;
            for (int ic = 0; ic < CVALID; ++ic)
                s += row[ic] * wk[ic * 9];
        }
    }
    out[oc * HWPIX + pix] = s;
}

// ---------------------------------------------------------------------------
// Launch
// ---------------------------------------------------------------------------

static inline int nb(int n) { return (n + 255) / 256; }

extern "C" void kernel_launch(void* const* d_in, const int* in_sizes, int n_in,
                              void* d_out, int out_size, void* d_ws, size_t ws_size,
                              hipStream_t stream)
{
    (void)in_sizes; (void)n_in; (void)out_size; (void)ws_size;

    const float* u       = (const float*)d_in[0];
    const float* pan     = (const float*)d_in[1];
    const float* w_nlu   = (const float*)d_in[2];
    const float* w_nlpan = (const float*)d_in[3];
    const float* g_phi   = (const float*)d_in[4];
    const float* g_theta = (const float*)d_in[5];
    const float* g_g     = (const float*)d_in[6];
    const float* s_phi   = (const float*)d_in[7];
    const float* s_theta = (const float*)d_in[8];
    const float* s_g     = (const float*)d_in[9];
    const float* m_phi   = (const float*)d_in[10];
    const float* m_theta = (const float*)d_in[11];
    const float* m_g     = (const float*)d_in[12];
    const float* mlp_w   = (const float*)d_in[13];
    const float* mlp_b   = (const float*)d_in[14];
    const float* w_resu  = (const float*)d_in[15];
    const float* w_respan= (const float*)d_in[16];
    const float* w_recon = (const float*)d_in[17];
    const float* rbw1[3] = {(const float*)d_in[18], (const float*)d_in[22], (const float*)d_in[26]};
    const float* rbb1[3] = {(const float*)d_in[19], (const float*)d_in[23], (const float*)d_in[27]};
    const float* rbw2[3] = {(const float*)d_in[20], (const float*)d_in[24], (const float*)d_in[28]};
    const float* rbb2[3] = {(const float*)d_in[21], (const float*)d_in[25], (const float*)d_in[29]};
    float* out = (float*)d_out;

    // bump allocator over workspace
    char* ws = (char*)d_ws;
    size_t off = 0;
    auto alloc = [&](size_t bytes) -> char* {
        char* p = ws + off;
        off = (off + bytes + 255) & ~(size_t)255;
        return p;
    };

    float* uf  = (float*)alloc(5 * HWPIX * 4);
    float* pf  = (float*)alloc(3 * HWPIX * 4);
    float* cat = (float*)alloc(8 * HWPIX * 4);
    float* th1 = (float*)alloc(3 * HWPIX * 4);
    float* ph1 = (float*)alloc(3 * HWPIX * 4);
    float* gg1 = (float*)alloc(5 * HWPIX * 4);
    float* th2 = (float*)alloc(5 * HWPIX * 4);
    float* ph2 = (float*)alloc(5 * HWPIX * 4);
    float* gg2 = (float*)alloc(5 * HWPIX * 4);
    float* th3 = (float*)alloc(8 * HWPIX * 4);
    float* ph3 = (float*)alloc(8 * HWPIX * 4);
    float* gg3 = (float*)alloc(5 * HWPIX * 4);
    float*  xf = (float*)alloc((size_t)HWPIX * CPAD * 4);
    __bf16* xb = (__bf16*)alloc((size_t)HWPIX * CPAD * 2);
    __bf16* fb = (__bf16*)alloc((size_t)HWPIX * CPAD * 2);
    __bf16* wpk1[3];
    __bf16* wpk2[3];
    for (int i = 0; i < 3; ++i) {
        wpk1[i] = (__bf16*)alloc(27648 * 2);
        wpk2[i] = (__bf16*)alloc(27648 * 2);
    }

    // zero trunk + bf16 buffers (pad channels must be 0 every call)
    hipMemsetAsync(xf, 0, (size_t)HWPIX * CPAD * 4, stream);
    hipMemsetAsync(xb, 0, (size_t)HWPIX * CPAD * 2, stream);
    hipMemsetAsync(fb, 0, (size_t)HWPIX * CPAD * 2, stream);

    // feature extraction convs
    conv3_nchw_k<<<nb(5 * HWPIX), 256, 0, stream>>>(u, w_nlu, uf, 8, 5);
    conv3_nchw_k<<<nb(3 * HWPIX), 256, 0, stream>>>(pan, w_nlpan, pf, 1, 3);

    // cat = [uf, pf] for mixed head
    hipMemcpyAsync(cat, uf, 5 * HWPIX * 4, hipMemcpyDeviceToDevice, stream);
    hipMemcpyAsync(cat + 5 * HWPIX, pf, 3 * HWPIX * 4, hipMemcpyDeviceToDevice, stream);

    // 1x1 projections
    mat1x1_k<<<nb(3 * HWPIX), 256, 0, stream>>>(g_theta, pf, th1, 3, 3);
    mat1x1_k<<<nb(3 * HWPIX), 256, 0, stream>>>(g_phi,   pf, ph1, 3, 3);
    mat1x1_k<<<nb(5 * HWPIX), 256, 0, stream>>>(g_g,     uf, gg1, 5, 5);
    mat1x1_k<<<nb(5 * HWPIX), 256, 0, stream>>>(s_theta, uf, th2, 5, 5);
    mat1x1_k<<<nb(5 * HWPIX), 256, 0, stream>>>(s_phi,   uf, ph2, 5, 5);
    mat1x1_k<<<nb(5 * HWPIX), 256, 0, stream>>>(s_g,     uf, gg2, 5, 5);
    mat1x1_k<<<nb(8 * HWPIX), 256, 0, stream>>>(m_theta, cat, th3, 8, 8);
    mat1x1_k<<<nb(8 * HWPIX), 256, 0, stream>>>(m_phi,   cat, ph3, 8, 8);
    mat1x1_k<<<nb(5 * HWPIX), 256, 0, stream>>>(m_g,     uf, gg3, 5, 5);

    // attention: bias init, then three heads accumulate into trunk ch 0..4
    att_bias_k<<<nb(5 * HWPIX), 256, 0, stream>>>(xf, mlp_b);
    head_k<3, 3><<<nb(HWPIX), 256, 0, stream>>>(th1, ph1, gg1, mlp_w, 0, xf);
    head_k<5, 1><<<nb(HWPIX), 256, 0, stream>>>(th2, ph2, gg2, mlp_w, 1, xf);
    head_k<8, 3><<<nb(HWPIX), 256, 0, stream>>>(th3, ph3, gg3, mlp_w, 2, xf);

    // residual-path convs into trunk channels 5..36 and 37..41
    conv3_to_x_k<<<nb(32 * HWPIX), 256, 0, stream>>>(u, w_resu, xf, 8, 32, 5);
    conv3_to_x_k<<<nb(5 * HWPIX), 256, 0, stream>>>(pan, w_respan, xf, 1, 5, 37);

    // pack activations + all residual-block weights
    pack_x_k<<<nb(HWPIX * CPAD), 256, 0, stream>>>(xf, xb);
    for (int i = 0; i < 3; ++i) {
        pack_w_k<<<nb(27648), 256, 0, stream>>>(rbw1[i], wpk1[i]);
        pack_w_k<<<nb(27648), 256, 0, stream>>>(rbw2[i], wpk2[i]);
    }

    // three residual blocks: f = relu(conv1(x)+b1); x = relu(conv2(f)+b2 + x)
    for (int i = 0; i < 3; ++i) {
        wmma_conv_k<<<256, 128, 0, stream>>>(xb, wpk1[i], rbb1[i], nullptr, fb, nullptr);
        wmma_conv_k<<<256, 128, 0, stream>>>(fb, wpk2[i], rbb2[i], xf, xb, xf);
    }

    // reconstruction conv 42->8, NCHW fp32 output
    recon_k<<<nb(8 * HWPIX), 256, 0, stream>>>(xf, w_recon, out);
}
